// MILModel9_7456063226208
// MI455X (gfx1250) — compile-verified, matrix operand order checked by reading, not run on hardware
//
#include <hip/hip_runtime.h>
#include <hip/hip_bf16.h>

typedef __attribute__((ext_vector_type(16))) _Float16 v16h;
typedef __attribute__((ext_vector_type(8)))  _Float16 h8;
typedef __attribute__((ext_vector_type(8)))  float    v8f;
typedef __attribute__((ext_vector_type(4)))  float    f32x4;

#define B_SZ 32
#define N_SZ 2048
#define VD 512
#define AD 128
#define SD 128
#define HD 64

__device__ __forceinline__ v8f wmma_f16(v16h a, v16h b, v8f c) {
    return __builtin_amdgcn_wmma_f32_16x16x32_f16(
        false, a, false, b, (short)0, c, false, false);
}

__device__ __forceinline__ float sigmoidf(float x) {
    return 1.0f / (1.0f + __expf(-x));
}

__device__ __forceinline__ void lds_fence() {
    asm volatile("s_wait_dscnt 0" ::: "memory");
}

// A-fragment (16x32, f16) from row-major fp32 global. Lane L<16: row L, K in
// {base..base+7} u {base+16..base+23}; lanes 16-31: base += 8. (ISA 7.12.2)
__device__ __forceinline__ v16h load_a_f32(const float* __restrict__ p, int ld,
                                           int kb, int lane) {
    const float* r = p + (size_t)(lane & 15) * ld + kb + ((lane >> 4) << 3);
    f32x4 x0 = *(const f32x4*)(r + 0);
    f32x4 x1 = *(const f32x4*)(r + 4);
    f32x4 y0 = *(const f32x4*)(r + 16);
    f32x4 y1 = *(const f32x4*)(r + 20);
    v16h a;
    a[0]  = (_Float16)x0[0]; a[1]  = (_Float16)x0[1];
    a[2]  = (_Float16)x0[2]; a[3]  = (_Float16)x0[3];
    a[4]  = (_Float16)x1[0]; a[5]  = (_Float16)x1[1];
    a[6]  = (_Float16)x1[2]; a[7]  = (_Float16)x1[3];
    a[8]  = (_Float16)y0[0]; a[9]  = (_Float16)y0[1];
    a[10] = (_Float16)y0[2]; a[11] = (_Float16)y0[3];
    a[12] = (_Float16)y1[0]; a[13] = (_Float16)y1[1];
    a[14] = (_Float16)y1[2]; a[15] = (_Float16)y1[3];
    return a;
}

// 16x32 fragment from row-major f16 LDS (row = lane&15 of [r0..r0+15]).
// Used for both B-fragments (row = output channel n) and A-fragments (row = m).
__device__ __forceinline__ v16h load_frag_h(const _Float16* w, int ld, int r0,
                                            int kb, int lane) {
    const _Float16* p = w + (size_t)(r0 + (lane & 15)) * ld + kb + ((lane >> 4) << 3);
    h8 b0 = *(const h8*)(p);
    h8 b1 = *(const h8*)(p + 16);
    v16h b;
#pragma unroll
    for (int i = 0; i < 8; ++i) { b[i] = b0[i]; b[i + 8] = b1[i]; }
    return b;
}

// ---------------- weight fp32 -> f16 conversion (runs once per call) --------
__global__ void cvt_weights(const float* __restrict__ vfc,
                            const float* __restrict__ afc,
                            const float* __restrict__ am,
                            const float* __restrict__ U,
                            const float* __restrict__ V,
                            _Float16* __restrict__ w16) {
    int i = blockIdx.x * 256 + threadIdx.x;
    float v;
    if (i < 65536)       v = vfc[i];
    else if (i < 81920)  v = afc[i - 65536];
    else if (i < 98304)  v = am[i - 81920];
    else if (i < 106496) v = U[i - 98304];
    else if (i < 114688) v = V[i - 106496];
    else return;
    w16[i] = (_Float16)v;
}

// ---------------- fused MIL forward -----------------------------------------
__global__ __launch_bounds__(256, 1)
void mil_fused(const float* __restrict__ vfeat, const float* __restrict__ afeat,
               const _Float16* __restrict__ w16,
               const float* __restrict__ vfc_b, const float* __restrict__ afc_b,
               const float* __restrict__ am_b, const float* __restrict__ U_b,
               const float* __restrict__ V_b, const float* __restrict__ W_w,
               const float* __restrict__ W_b,
               float* __restrict__ out_s, float* __restrict__ zfeat) {
    __shared__ _Float16 w_afc[SD * SD];      // 32 KB
    __shared__ _Float16 w_am[SD * SD];       // 32 KB
    __shared__ _Float16 w_U[HD * SD];        // 16 KB
    __shared__ _Float16 w_V[HD * SD];        // 16 KB
    __shared__ _Float16 bchunk[SD * 32];     // 8 KB vfc_w K-chunk
    __shared__ _Float16 ax_s[8][16 * SD];    // 32 KB per-wave ax tiles
    __shared__ _Float16 ff_s[8][16 * SD];    // 32 KB per-wave ffeat tiles
    __shared__ float s_s[8][16];
    __shared__ float b_vfc[SD], b_afc[SD], b_am[SD], b_U[HD], b_V[HD], wWs[HD];
    __shared__ float wBs;

    const int tid  = threadIdx.x;
    const int wave = tid >> 5;
    const int lane = tid & 31;
    const int wg   = blockIdx.x;      // 512 workgroups
    const int b    = wg >> 4;         // 16 WGs span one batch (2048/128)
    const int row0 = (wg & 15) * 128;

    const _Float16* g_vfc = w16;
    const _Float16* g_afc = w16 + 65536;
    const _Float16* g_am  = g_afc + 16384;
    const _Float16* g_U   = g_am + 16384;
    const _Float16* g_V   = g_U + 8192;

    // Stage small weights + biases into LDS.
    for (int i = tid; i < 2048; i += 256) ((h8*)w_afc)[i] = ((const h8*)g_afc)[i];
    for (int i = tid; i < 2048; i += 256) ((h8*)w_am)[i]  = ((const h8*)g_am)[i];
    for (int i = tid; i < 1024; i += 256) ((h8*)w_U)[i]   = ((const h8*)g_U)[i];
    for (int i = tid; i < 1024; i += 256) ((h8*)w_V)[i]   = ((const h8*)g_V)[i];
    if (tid < SD) { b_vfc[tid] = vfc_b[tid]; b_afc[tid] = afc_b[tid]; b_am[tid] = am_b[tid]; }
    if (tid < HD) { b_U[tid] = U_b[tid]; b_V[tid] = V_b[tid]; wWs[tid] = W_w[tid]; }
    if (tid == 0) wBs = W_b[0];
    __syncthreads();

    const v8f vzero = {0.f, 0.f, 0.f, 0.f, 0.f, 0.f, 0.f, 0.f};
    const int mhi = (lane >> 4) << 3;   // D-layout row offset for this lane half
    const int nlo = lane & 15;          // D-layout column within a 16-wide tile

    // ---- GEMM1: vo1 = vfeat @ vfc_w^T  (K=512, streamed B through LDS) ----
    const float* va = vfeat + ((size_t)b * N_SZ + row0 + wave * 16) * VD;
    v8f acc1[8];
#pragma unroll
    for (int t = 0; t < 8; ++t) acc1[t] = vzero;

    for (int kc = 0; kc < VD; kc += 32) {
        {   // cooperative stage of vfc_w f16 chunk [128 n][32 k]
            int n  = tid >> 1;
            int kh = (tid & 1) * 16;
            *(h8*)&bchunk[n * 32 + kh]     = *(const h8*)&g_vfc[n * VD + kc + kh];
            *(h8*)&bchunk[n * 32 + kh + 8] = *(const h8*)&g_vfc[n * VD + kc + kh + 8];
        }
        if (kc + 32 < VD) __builtin_prefetch(va + kc + 32, 0, 1);
        __syncthreads();
        v16h a = load_a_f32(va, VD, kc, lane);
#pragma unroll
        for (int t = 0; t < 8; ++t)
            acc1[t] = wmma_f16(a, load_frag_h(bchunk, 32, t * 16, 0, lane), acc1[t]);
        __syncthreads();
    }

    // ---- GEMM2: ax = relu(afeat @ afc_w^T)  (K=128) ----
    const float* aa = afeat + ((size_t)b * N_SZ + row0 + wave * 16) * AD;
    v8f acc2[8];
#pragma unroll
    for (int t = 0; t < 8; ++t) acc2[t] = vzero;
#pragma unroll
    for (int kc = 0; kc < AD; kc += 32) {
        v16h a = load_a_f32(aa, AD, kc, lane);
#pragma unroll
        for (int t = 0; t < 8; ++t)
            acc2[t] = wmma_f16(a, load_frag_h(w_afc, SD, t * 16, kc, lane), acc2[t]);
    }
#pragma unroll
    for (int t = 0; t < 8; ++t) {
        int n = t * 16 + nlo;
        float bb = b_afc[n];
#pragma unroll
        for (int r = 0; r < 8; ++r) {
            float vx = acc2[t][r] + bb;
            vx = vx > 0.f ? vx : 0.f;
            ax_s[wave][(r + mhi) * SD + n] = (_Float16)vx;
        }
    }
    lds_fence();  // same-wave D-store -> A-load RAW through LDS

    // ---- GEMM3: gate = sigmoid(ax @ am_w^T); ffeat = vo1*gate + vo1 ----
    v8f acc3[8];
#pragma unroll
    for (int t = 0; t < 8; ++t) acc3[t] = vzero;
#pragma unroll
    for (int kc = 0; kc < SD; kc += 32) {
        v16h a = load_frag_h(ax_s[wave], SD, 0, kc, lane);
#pragma unroll
        for (int t = 0; t < 8; ++t)
            acc3[t] = wmma_f16(a, load_frag_h(w_am, SD, t * 16, kc, lane), acc3[t]);
    }
#pragma unroll
    for (int t = 0; t < 8; ++t) {
        int n = t * 16 + nlo;
        float bg = b_am[n], bv = b_vfc[n];
#pragma unroll
        for (int r = 0; r < 8; ++r) {
            float vo = acc1[t][r] + bv;
            vo = vo > 0.f ? vo : 0.f;
            float g  = sigmoidf(acc3[t][r] + bg);
            ff_s[wave][(r + mhi) * SD + n] = (_Float16)(vo * g + vo);
        }
    }
    lds_fence();

    // ---- GEMM4/5: v = relu(ffeat@V^T), u = sigmoid(ffeat@U^T)  (N=64) ----
    v8f accv[4], accu[4];
#pragma unroll
    for (int t = 0; t < 4; ++t) { accv[t] = vzero; accu[t] = vzero; }
#pragma unroll
    for (int kc = 0; kc < SD; kc += 32) {
        v16h a = load_frag_h(ff_s[wave], SD, 0, kc, lane);
#pragma unroll
        for (int t = 0; t < 4; ++t) {
            accv[t] = wmma_f16(a, load_frag_h(w_V, SD, t * 16, kc, lane), accv[t]);
            accu[t] = wmma_f16(a, load_frag_h(w_U, SD, t * 16, kc, lane), accu[t]);
        }
    }

    // ---- score head: s[m] = sum_h relu(v)*sigmoid(u)*W[h] + W_b ----
#pragma unroll
    for (int r = 0; r < 8; ++r) {
        float p = 0.f;
#pragma unroll
        for (int t = 0; t < 4; ++t) {
            int h = t * 16 + nlo;
            float vv = accv[t][r] + b_V[h];
            vv = vv > 0.f ? vv : 0.f;
            float uu = sigmoidf(accu[t][r] + b_U[h]);
            p += vv * uu * wWs[h];
        }
        // reduce over the 16 lanes holding the same m (xor<16 keeps halves apart)
        p += __shfl_xor(p, 1);
        p += __shfl_xor(p, 2);
        p += __shfl_xor(p, 4);
        p += __shfl_xor(p, 8);
        if (nlo == 0) {
            int m = r + mhi;
            float sv = p + wBs;
            s_s[wave][m] = sv;
            out_s[(size_t)b * N_SZ + row0 + wave * 16 + m] = sv;
        }
    }
    __syncthreads();

    // ---- zfeat partial: z[b,k] += sum over this WG's 128 rows of s*ffeat ----
    if (tid < SD) {
        float z = 0.f;
#pragma unroll 2
        for (int w = 0; w < 8; ++w)
#pragma unroll
            for (int m = 0; m < 16; ++m)
                z += s_s[w][m] * (float)ff_s[w][m * SD + tid];
        atomicAdd(&zfeat[b * SD + tid], z);
    }
}

// ---------------- final classifier: logits = zfeat @ cls_w^T + cls_b -------
__global__ void mil_head(const float* __restrict__ zfeat,
                         const float* __restrict__ cls_w,
                         const float* __restrict__ cls_b,
                         float* __restrict__ out_logits) {
    int t = threadIdx.x;
    if (t >= B_SZ * 2) return;
    int bb = t >> 1, c = t & 1;
    const float* z = zfeat + bb * SD;
    const float* w = cls_w + c * SD;
    float acc = 0.f;
#pragma unroll 4
    for (int k = 0; k < SD; ++k) acc += z[k] * w[k];
    out_logits[bb * 2 + c] = acc + cls_b[c];
}

extern "C" void kernel_launch(void* const* d_in, const int* in_sizes, int n_in,
                              void* d_out, int out_size, void* d_ws, size_t ws_size,
                              hipStream_t stream) {
    const float* vfeat = (const float*)d_in[0];
    const float* afeat = (const float*)d_in[1];
    const float* vfc_w = (const float*)d_in[2];
    const float* vfc_b = (const float*)d_in[3];
    const float* afc_w = (const float*)d_in[4];
    const float* afc_b = (const float*)d_in[5];
    const float* am_w  = (const float*)d_in[6];
    const float* am_b  = (const float*)d_in[7];
    const float* U_w   = (const float*)d_in[8];
    const float* U_b   = (const float*)d_in[9];
    const float* V_w   = (const float*)d_in[10];
    const float* V_b   = (const float*)d_in[11];
    const float* W_w   = (const float*)d_in[12];
    const float* W_b   = (const float*)d_in[13];
    const float* cls_w = (const float*)d_in[14];
    const float* cls_b = (const float*)d_in[15];

    float*     zfeat = (float*)d_ws;                          // [32,128]
    _Float16*  w16   = (_Float16*)((char*)d_ws + 16384);      // 114688 halves
    float*     out_s = (float*)d_out;                         // [32,2048]
    float*     out_l = out_s + (size_t)B_SZ * N_SZ;           // [32,2]

    hipMemsetAsync(zfeat, 0, B_SZ * SD * sizeof(float), stream);
    cvt_weights<<<448, 256, 0, stream>>>(vfc_w, afc_w, am_w, U_w, V_w, w16);
    mil_fused<<<(B_SZ * N_SZ) / 128, 256, 0, stream>>>(
        vfeat, afeat, w16, vfc_b, afc_b, am_b, U_b, V_b, W_w, W_b, out_s, zfeat);
    mil_head<<<1, 64, 0, stream>>>(zfeat, cls_w, cls_b, out_l);
}
